// CellDrugAttentionModel_63986422776352
// MI455X (gfx1250) — matrix operand order repaired and checked by reading, not compiled, not run on hardware
//
#include <hip/hip_runtime.h>
#include <hip/hip_bf16.h>

typedef __bf16 bf16;
typedef bf16  v16bf __attribute__((ext_vector_type(16)));
typedef float v8f   __attribute__((ext_vector_type(8)));
typedef int   vs4i  __attribute__((vector_size(16)));

#define BM 128
#define BN 64
#define BK 32
#define ASTR 40   // BK + 8 pad (80 bytes/row: keeps 16B alignment, spreads banks)
#define MAXLEN 128

#define AS1 __attribute__((address_space(1)))
#define AS3 __attribute__((address_space(3)))

// async 16-byte global -> LDS copy (per lane), tracked by ASYNCcnt
#define ASYNC_CP16(ldst, gsrc)                                                      \
    __builtin_amdgcn_global_load_async_to_lds_b128(                                 \
        (AS1 vs4i*)(gsrc), (AS3 vs4i*)(ldst), 0, 0)

union Frag { v16bf v; uint4 q[2]; };

// ---------------- fp32 -> bf16 conversion (vector-4) ----------------
__global__ void cvt4_kernel(const float4* __restrict__ s, ushort4* __restrict__ d, long n4) {
    long i = (long)blockIdx.x * blockDim.x + threadIdx.x;
    if (i >= n4) return;
    float4 f = s[i];
    ushort4 o;
    o.x = __builtin_bit_cast(unsigned short, (bf16)f.x);
    o.y = __builtin_bit_cast(unsigned short, (bf16)f.y);
    o.z = __builtin_bit_cast(unsigned short, (bf16)f.z);
    o.w = __builtin_bit_cast(unsigned short, (bf16)f.w);
    d[i] = o;
}

// ---------------- segment starts via lower_bound (guide is sorted) ----------------
__global__ void compute_starts_kernel(const int* __restrict__ guide, int N, int B,
                                      int* __restrict__ starts) {
    int b = blockIdx.x * blockDim.x + threadIdx.x;
    if (b > B) return;
    if (b == B) { starts[B] = N; return; }
    int lo = 0, hi = N;
    while (lo < hi) {
        int mid = (lo + hi) >> 1;
        if (guide[mid] < b) lo = mid + 1; else hi = mid;
    }
    starts[b] = lo;
}

// ---------------- generic bf16 WMMA GEMM: C = (A @ Bw^T + bias) * scale ----------------
// A: [M,K] bf16 row-major; Bw: [Nd,K] bf16 row-major (B transposed).
// Block tile 128x64 with 128 threads = 4 waves; wave w computes a 32x64 tile
// (2 A-frags x 4 B-frags -> 8 WMMAs per K-step). LDS is double-buffered and filled
// with GLOBAL_LOAD_ASYNC_TO_LDS_B128 so tile k+1 streams in while tile k is in the XDL.
__global__ void __launch_bounds__(128)
gemm_bf16_nt(const bf16* __restrict__ A, const bf16* __restrict__ Bw,
             const float* __restrict__ bias,
             float* __restrict__ Cf, int ldcf,
             bf16*  __restrict__ Cb, int ldcb,
             int M, int Nd, int K, float scale) {
    __shared__ __align__(16) bf16 As[2][BM * ASTR];
    __shared__ __align__(16) bf16 Bs[2][BN * ASTR];

    const int bm = blockIdx.x * BM;
    const int bn = blockIdx.y * BN;
    const int t = threadIdx.x;
    const int wave = t >> 5, lane = t & 31;
    const int row16 = lane & 15, hi = lane >> 4;

    // global staging pointers (row clamped if OOB: garbage only lands in rows >= M,
    // which the epilogue never stores)
    const int arow_g = (bm + t) < M ? (bm + t) : 0;
    const bf16* aptr = A + (size_t)arow_g * K;                       // 64B per K-step
    const bf16* bptr = Bw + (size_t)(bn + (t >> 1)) * K + (t & 1) * 16; // 32B per K-step
    bf16* adst = &As[0][t * ASTR];
    bf16* bdst = &Bs[0][(t >> 1) * ASTR + (t & 1) * 16];
    const int abufoff = BM * ASTR;   // elements between As[0] and As[1]
    const int bbufoff = BN * ASTR;

    // fragment LDS read offsets (element offsets inside one buffer)
    const int a_rd0 = (wave * 32 + row16) * ASTR + hi * 8;
    const int a_rd1 = (wave * 32 + 16 + row16) * ASTR + hi * 8;
    const int b_rd  = row16 * ASTR + hi * 8;

    v8f acc[2][4] = {};

    auto stage_async = [&](int buf, int k0) {
        const bf16* ap = aptr + k0;
        bf16* ad = adst + buf * abufoff;
        ASYNC_CP16(ad +  0, ap +  0);
        ASYNC_CP16(ad +  8, ap +  8);
        ASYNC_CP16(ad + 16, ap + 16);
        ASYNC_CP16(ad + 24, ap + 24);
        const bf16* bp = bptr + k0;
        bf16* bd = bdst + buf * bbufoff;
        ASYNC_CP16(bd + 0, bp + 0);
        ASYNC_CP16(bd + 8, bp + 8);
    };

    const int nsteps = K / BK;
    stage_async(0, 0);
    __builtin_amdgcn_s_wait_asynccnt(0);
    __syncthreads();

    for (int s = 0; s < nsteps; ++s) {
        const int buf = s & 1;
        if (s + 1 < nsteps) stage_async(buf ^ 1, (s + 1) * BK);

        const bf16* Ab = &As[0][0] + buf * abufoff;
        const bf16* Bb = &Bs[0][0] + buf * bbufoff;
        Frag af[2], bfr[4];
        af[0].q[0] = *(const uint4*)(Ab + a_rd0);
        af[0].q[1] = *(const uint4*)(Ab + a_rd0 + 16);
        af[1].q[0] = *(const uint4*)(Ab + a_rd1);
        af[1].q[1] = *(const uint4*)(Ab + a_rd1 + 16);
#pragma unroll
        for (int nt = 0; nt < 4; ++nt) {
            bfr[nt].q[0] = *(const uint4*)(Bb + b_rd + nt * 16 * ASTR);
            bfr[nt].q[1] = *(const uint4*)(Bb + b_rd + nt * 16 * ASTR + 16);
        }
#pragma unroll
        for (int mi = 0; mi < 2; ++mi)
#pragma unroll
            for (int nt = 0; nt < 4; ++nt)
                acc[mi][nt] = __builtin_amdgcn_wmma_f32_16x16x32_bf16(
                    false, af[mi].v, false, bfr[nt].v, (short)0, acc[mi][nt], false, false);

        if (s + 1 < nsteps) {
            __builtin_amdgcn_s_wait_asynccnt(0);
            __syncthreads();
        }
    }

    // epilogue: lane<16 -> rows 0..7 col lane; lane>=16 -> rows 8..15 col lane-16
#pragma unroll
    for (int mi = 0; mi < 2; ++mi) {
        int mbase = bm + wave * 32 + mi * 16 + hi * 8;
#pragma unroll
        for (int nt = 0; nt < 4; ++nt) {
            int col = bn + nt * 16 + row16;
            float bv = bias ? bias[col] : 0.f;
#pragma unroll
            for (int j = 0; j < 8; ++j) {
                int rowg = mbase + j;
                if (rowg >= M) break;
                float vout = (acc[mi][nt][j] + bv) * scale;
                if (Cf) Cf[(size_t)rowg * ldcf + col] = vout;
                if (Cb) Cb[(size_t)rowg * ldcb + col] = (bf16)vout;
            }
        }
    }
}

// ---------------- per-graph single-query attention ----------------
__global__ void __launch_bounds__(256)
attn_kernel(const float* __restrict__ qptr,   // [B,1024] already scaled by 1/8
            const bf16*  __restrict__ Kp,     // [N,1024]
            const bf16*  __restrict__ Vp,     // [N,1024]
            const int*   __restrict__ starts, // [B+1]
            bf16* __restrict__ outp) {        // [B,1024] bf16
    const int b = blockIdx.x;
    __shared__ float qs[1024];
    __shared__ float sc[16][MAXLEN];
    const int t = threadIdx.x;

    for (int i = t; i < 1024; i += 256) qs[i] = qptr[(size_t)b * 1024 + i];
    __syncthreads();

    const int s0 = starts[b];
    int m = starts[b + 1] - s0;
    if (m > MAXLEN) m = MAXLEN;   // reference's to_dense_batch drops overflow

    const int wave = t >> 5, lane = t & 31;
    // scores: wave handles node j; lane pair (2 lanes per head) splits the 64-dim dot
    {
        int h = lane >> 1, half = (lane & 1) * 32;
        for (int j = wave; j < m; j += 8) {
            const bf16* krow = Kp + (size_t)(s0 + j) * 1024 + h * 64 + half;
            float acc = 0.f;
#pragma unroll
            for (int d = 0; d < 32; ++d) acc += qs[h * 64 + half + d] * (float)krow[d];
            acc += __shfl_xor(acc, 1, 32);
            if ((lane & 1) == 0) sc[h][j] = acc;
        }
    }
    __syncthreads();

    // softmax over valid keys: 16 threads per head (groups of 16 lanes)
    {
        int h = t >> 4, r = t & 15;
        float mx = -1e30f;
        for (int j = r; j < m; j += 16) mx = fmaxf(mx, sc[h][j]);
        for (int o = 8; o; o >>= 1) mx = fmaxf(mx, __shfl_xor(mx, o, 16));
        float sum = 0.f;
        for (int j = r; j < m; j += 16) { float e = __expf(sc[h][j] - mx); sc[h][j] = e; sum += e; }
        for (int o = 8; o; o >>= 1) sum += __shfl_xor(sum, o, 16);
        float inv = (m > 0) ? 1.f / sum : 0.f;
        for (int j = r; j < m; j += 16) sc[h][j] *= inv;
    }
    __syncthreads();

    // out[h,d] = sum_j p[h,j] * V[s0+j, h*64+d]; coalesced in d
#pragma unroll
    for (int rep = 0; rep < 4; ++rep) {
        int idx = t + rep * 256;       // 0..1023
        int h = idx >> 6;
        float acc = 0.f;
        for (int j = 0; j < m; ++j)
            acc += sc[h][j] * (float)Vp[(size_t)(s0 + j) * 1024 + idx];
        outp[(size_t)b * 1024 + idx] = (bf16)acc;
    }
}

// ---------------- concat: d_out[:, 1024:2048] = cell ----------------
__global__ void concat_cell_kernel(const float* __restrict__ cell, float* __restrict__ out, int n) {
    int i = blockIdx.x * blockDim.x + threadIdx.x;
    if (i >= n) return;
    int b = i >> 10, c = i & 1023;
    out[(size_t)b * 2048 + 1024 + c] = cell[i];
}

extern "C" void kernel_launch(void* const* d_in, const int* in_sizes, int n_in,
                              void* d_out, int out_size, void* d_ws, size_t ws_size,
                              hipStream_t stream) {
    const float* x    = (const float*)d_in[0];
    const float* cell = (const float*)d_in[1];
    const float* Wq   = (const float*)d_in[2];
    const float* bq   = (const float*)d_in[3];
    const float* Wk   = (const float*)d_in[4];
    const float* bk   = (const float*)d_in[5];
    const float* Wv   = (const float*)d_in[6];
    const float* bv   = (const float*)d_in[7];
    const float* Wo   = (const float*)d_in[8];
    const float* bo   = (const float*)d_in[9];
    const float* Wc   = (const float*)d_in[10];
    const float* bc   = (const float*)d_in[11];
    const int*   guide = (const int*)d_in[12];

    const int DD = 256, E = 1024, DC = 1024;
    const int N = in_sizes[0] / DD;   // 32768
    const int B = in_sizes[1] / DC;   // 512

    char* ws = (char*)d_ws;
    size_t off = 0;
    auto alloc = [&](size_t bytes) -> char* {
        char* p = ws + off;
        off += (bytes + 255) & ~(size_t)255;
        return p;
    };
    bf16* x_bf    = (bf16*)alloc((size_t)N * DD * 2);
    bf16* cell_bf = (bf16*)alloc((size_t)B * DC * 2);
    bf16* wc_bf   = (bf16*)alloc((size_t)E * DC * 2);
    bf16* wq_bf   = (bf16*)alloc((size_t)E * E * 2);
    bf16* wk_bf   = (bf16*)alloc((size_t)E * DD * 2);
    bf16* wv_bf   = (bf16*)alloc((size_t)E * DD * 2);
    bf16* wo_bf   = (bf16*)alloc((size_t)E * E * 2);
    bf16* cq_bf   = (bf16*)alloc((size_t)B * E * 2);
    float* q_f    = (float*)alloc((size_t)B * E * 4);
    bf16* K_bf    = (bf16*)alloc((size_t)N * E * 2);
    bf16* V_bf    = (bf16*)alloc((size_t)N * E * 2);
    bf16* ao_bf   = (bf16*)alloc((size_t)B * E * 2);
    int*  starts  = (int*)alloc((size_t)(B + 1) * 4);

    auto cvt = [&](const float* s, bf16* d, long n) {
        long n4 = n / 4;
        cvt4_kernel<<<dim3((unsigned)((n4 + 255) / 256)), dim3(256), 0, stream>>>(
            (const float4*)s, (ushort4*)d, n4);
    };
    cvt(x, x_bf, (long)N * DD);
    cvt(cell, cell_bf, (long)B * DC);
    cvt(Wc, wc_bf, (long)E * DC);
    cvt(Wq, wq_bf, (long)E * E);
    cvt(Wk, wk_bf, (long)E * DD);
    cvt(Wv, wv_bf, (long)E * DD);
    cvt(Wo, wo_bf, (long)E * E);

    compute_starts_kernel<<<dim3((unsigned)((B + 1 + 255) / 256)), dim3(256), 0, stream>>>(
        guide, N, B, starts);

    const dim3 blk(128);
    // cq = cell @ Wc^T + bc  -> bf16
    gemm_bf16_nt<<<dim3((B + BM - 1) / BM, E / BN), blk, 0, stream>>>(
        cell_bf, wc_bf, bc, nullptr, 0, cq_bf, E, B, E, DC, 1.f);
    // q = (cq @ Wq^T + bq) * 1/sqrt(64) -> fp32
    gemm_bf16_nt<<<dim3((B + BM - 1) / BM, E / BN), blk, 0, stream>>>(
        cq_bf, wq_bf, bq, q_f, E, nullptr, 0, B, E, E, 0.125f);
    // K = X @ Wk^T + bk -> bf16
    gemm_bf16_nt<<<dim3((N + BM - 1) / BM, E / BN), blk, 0, stream>>>(
        x_bf, wk_bf, bk, nullptr, 0, K_bf, E, N, E, DD, 1.f);
    // V = X @ Wv^T + bv -> bf16
    gemm_bf16_nt<<<dim3((N + BM - 1) / BM, E / BN), blk, 0, stream>>>(
        x_bf, wv_bf, bv, nullptr, 0, V_bf, E, N, E, DD, 1.f);
    // per-graph attention
    attn_kernel<<<dim3(B), dim3(256), 0, stream>>>(q_f, K_bf, V_bf, starts, ao_bf);
    // out = attn_out @ Wo^T + bo -> fp32 into d_out columns [0,1024), row stride 2048
    gemm_bf16_nt<<<dim3((B + BM - 1) / BM, E / BN), blk, 0, stream>>>(
        ao_bf, wo_bf, bo, (float*)d_out, E + DC, nullptr, 0, B, E, E, 1.f);
    // d_out columns [1024,2048) = cell
    concat_cell_kernel<<<dim3((B * DC + 255) / 256), dim3(256), 0, stream>>>(
        cell, (float*)d_out, B * DC);
}
